// DynamicStripBlock_79164837199994
// MI455X (gfx1250) — matrix operand order repaired and checked by reading, not compiled, not run on hardware
//
#include <hip/hip_runtime.h>
#include <hip/hip_bf16.h>

typedef __attribute__((ext_vector_type(16))) _Float16 v16h;
typedef __attribute__((ext_vector_type(8)))  float    v8f;

#define B_  8
#define C_  256
#define HW_ 96
#define K_  19
#define A_  64
#define NH_ 8

union Frag16 { unsigned u[8]; v16h h; };

__device__ __forceinline__ float gelu_exact(float t) {
  return 0.5f * t * (1.f + erff(t * 0.70710678118654752f));
}

// ---------------------------------------------------------------------------
// Kernel 1: global average pool. One block per (b,c) plane of 96*96 floats.
// ---------------------------------------------------------------------------
__global__ __launch_bounds__(256)
void pool_kernel(const float* __restrict__ x, float* __restrict__ pooled) {
  const int bc = blockIdx.x;
  const float* xp = x + (size_t)bc * (HW_ * HW_);
  float s = 0.f;
  for (int i = threadIdx.x; i < HW_ * HW_; i += 256) s += xp[i];
  __shared__ float red[256];
  red[threadIdx.x] = s;
  __syncthreads();
  for (int st = 128; st > 0; st >>= 1) {
    if (threadIdx.x < st) red[threadIdx.x] += red[threadIdx.x + st];
    __syncthreads();
  }
  if (threadIdx.x == 0) pooled[bc] = red[0] * (1.f / (HW_ * HW_));
}

// ---------------------------------------------------------------------------
// Kernel 2: content MLP: hid = gelu(pooled@w1^T+b1); mod = hid@w2^T+b2
// ---------------------------------------------------------------------------
__global__ __launch_bounds__(256)
void mlp_kernel(const float* __restrict__ pooled,
                const float* __restrict__ w1, const float* __restrict__ b1,
                const float* __restrict__ w2, const float* __restrict__ b2,
                float* __restrict__ modout) {
  __shared__ float hid[B_ * 64];
  const int tid = threadIdx.x;
  for (int idx = tid; idx < B_ * 64; idx += 256) {
    int b = idx >> 6, j = idx & 63;
    const float* pb = pooled + b * C_;
    const float* wr = w1 + j * C_;
    float s = b1[j];
    for (int d = 0; d < C_; ++d) s = fmaf(pb[d], wr[d], s);
    hid[idx] = gelu_exact(s);
  }
  __syncthreads();
  for (int idx = tid; idx < B_ * 192; idx += 256) {
    int b = idx / 192, r = idx % 192;
    const float* hb = hid + b * 64;
    const float* wr = w2 + r * 64;
    float s = b2[r];
    for (int d = 0; d < 64; ++d) s = fmaf(hb[d], wr[d], s);
    modout[idx] = s;   // layout [b][3][64]
  }
}

// ---------------------------------------------------------------------------
// Kernel 3: dynamic-kernel MHA. One wave per (s,b,c) instance, 2 waves/block.
// In/out projections are WMMA f16->f32 GEMMs; attention core is VALU.
// ---------------------------------------------------------------------------
__global__ __launch_bounds__(64)
void dyn_mha_kernel(const float* __restrict__ mod,
                    const float* __restrict__ wp_w, const float* __restrict__ wp_b,
                    const float* __restrict__ in_w, const float* __restrict__ in_b,
                    const float* __restrict__ out_w, const float* __restrict__ out_b,
                    const float* __restrict__ base_h, const float* __restrict__ base_v,
                    float* __restrict__ dyn) {
  __shared__ _Float16 Wlds[64 * 64];        // staged weight (row = out idx, col = k)
  __shared__ _Float16 Xbuf[2][32 * 64];     // activations (reused as O)
  __shared__ _Float16 QH[2][32 * 64];
  __shared__ _Float16 KH[2][32 * 64];
  __shared__ _Float16 VH[2][32 * 64];
  __shared__ float bval[2][32];
  __shared__ float dynacc[2][32];
  __shared__ float wpwL[64], wpbL[64];

  const int tid  = threadIdx.x;
  const int w    = tid >> 5;       // wave in block
  const int lane = tid & 31;
  const int g    = lane >> 4;      // lane half-group
  const int ln   = lane & 15;

  const int inst = blockIdx.x * 2 + w;     // 0..4095
  const int s  = inst >> 11;               // 0=h, 1=v
  const int bc = inst & 2047;
  const int b  = bc >> 8;
  const int c  = bc & 255;
  const float* basep = (s == 0 ? base_h : base_v) + c * K_;
  const float* modb  = mod + (size_t)b * 3 * 64;

  if (tid < 64) { wpwL[tid] = wp_w[tid]; wpbL[tid] = wp_b[tid]; }
  bval[w][lane] = (lane < K_) ? basep[lane] : 0.f;
  const float bm = (lane < K_) ? basep[lane] : 0.f;

  _Float16* Xw = &Xbuf[w][0];

  // -------- three input projections via WMMA --------
  for (int p = 0; p < 3; ++p) {
    __syncthreads();                                     // Wlds free + staging visible
    for (int i = tid; i < 4096; i += 64) Wlds[i] = (_Float16)in_w[p * 4096 + i];
    const float* modp = modb + p * 64;
    #pragma unroll
    for (int d = 0; d < 64; ++d) {
      float xv = (lane < K_) ? (bm * wpwL[d] + wpbL[d] + modp[d]) : 0.f;
      Xw[lane * 64 + d] = (_Float16)xv;
    }
    __syncthreads();
    _Float16* dst = (p == 0 ? &QH[w][0] : (p == 1 ? &KH[w][0] : &VH[w][0]));
    const float* inbp = in_b + p * 64;
    #pragma unroll
    for (int Mt = 0; Mt < 32; Mt += 16) {
      const unsigned* xrow = (const unsigned*)(Xw + (Mt + ln) * 64);
      #pragma unroll
      for (int Nt = 0; Nt < 64; Nt += 16) {
        v8f acc = {0.f, 0.f, 0.f, 0.f, 0.f, 0.f, 0.f, 0.f};
        const int e = Nt + ln;
        const unsigned* wrow = (const unsigned*)(&Wlds[e * 64]);
        #pragma unroll
        for (int Kc = 0; Kc < 64; Kc += 32) {
          Frag16 af, bf;
          #pragma unroll
          for (int v = 0; v < 8; ++v) {
            int kb = Kc + g * 16 + v * 2;
            af.u[v] = xrow[kb >> 1];
          }
          const int kbase = (Kc + g * 16) >> 1;
          #pragma unroll
          for (int j = 0; j < 8; ++j) bf.u[j] = wrow[kbase + j];
          acc = __builtin_amdgcn_wmma_f32_16x16x32_f16(
              false, af.h, false, bf.h, (short)0, acc, false, false);
        }
        const float be = inbp[e];
        #pragma unroll
        for (int r = 0; r < 8; ++r) {
          int row = Mt + r + 8 * g;
          dst[row * 64 + e] = (_Float16)(acc[r] + be);
        }
      }
    }
  }
  __syncthreads();

  // -------- attention core (VALU): tasks = (l, head) --------
  _Float16* Ow = Xw;                       // reuse X buffer for attention output
  if (lane >= K_) {
    #pragma unroll
    for (int d = 0; d < 64; ++d) Ow[lane * 64 + d] = (_Float16)0.f;
  }
  const _Float16* QHw = &QH[w][0];
  const _Float16* KHw = &KH[w][0];
  const _Float16* VHw = &VH[w][0];
  for (int it = 0; it < 5; ++it) {
    int t = lane + it * 32;
    if (t < K_ * NH_) {
      int l = t >> 3, h = t & 7;
      float q[8];
      #pragma unroll
      for (int d = 0; d < 8; ++d) q[d] = (float)QHw[l * 64 + h * 8 + d];
      float sc[K_];
      float mx = -1e30f;
      #pragma unroll
      for (int m = 0; m < K_; ++m) {
        float sv = 0.f;
        #pragma unroll
        for (int d = 0; d < 8; ++d) sv = fmaf(q[d], (float)KHw[m * 64 + h * 8 + d], sv);
        sv *= 0.35355339059327f;           // 1/sqrt(8)
        sc[m] = sv;
        mx = fmaxf(mx, sv);
      }
      float ssum = 0.f;
      #pragma unroll
      for (int m = 0; m < K_; ++m) { float e = __expf(sc[m] - mx); sc[m] = e; ssum += e; }
      const float inv = 1.f / ssum;
      #pragma unroll
      for (int d = 0; d < 8; ++d) {
        float o = 0.f;
        #pragma unroll
        for (int m = 0; m < K_; ++m) o = fmaf(sc[m], (float)VHw[m * 64 + h * 8 + d], o);
        Ow[l * 64 + h * 8 + d] = (_Float16)(o * inv);
      }
    }
  }
  __syncthreads();
  for (int i = tid; i < 4096; i += 64) Wlds[i] = (_Float16)out_w[i];
  __syncthreads();

  // -------- output projection (WMMA) + (res + wsp).mean(-1) epilogue --------
  #pragma unroll
  for (int Mt = 0; Mt < 32; Mt += 16) {
    const unsigned* xrow = (const unsigned*)(Ow + (Mt + ln) * 64);
    float rsum[8] = {0.f, 0.f, 0.f, 0.f, 0.f, 0.f, 0.f, 0.f};
    #pragma unroll
    for (int Nt = 0; Nt < 64; Nt += 16) {
      v8f acc = {0.f, 0.f, 0.f, 0.f, 0.f, 0.f, 0.f, 0.f};
      const int a = Nt + ln;
      const unsigned* wrow = (const unsigned*)(&Wlds[a * 64]);
      #pragma unroll
      for (int Kc = 0; Kc < 64; Kc += 32) {
        Frag16 af, bf;
        #pragma unroll
        for (int v = 0; v < 8; ++v) {
          int kb = Kc + g * 16 + v * 2;
          af.u[v] = xrow[kb >> 1];
        }
        const int kbase = (Kc + g * 16) >> 1;
        #pragma unroll
        for (int j = 0; j < 8; ++j) bf.u[j] = wrow[kbase + j];
        acc = __builtin_amdgcn_wmma_f32_16x16x32_f16(
            false, af.h, false, bf.h, (short)0, acc, false, false);
      }
      const float ob = out_b[a];
      #pragma unroll
      for (int r = 0; r < 8; ++r) {
        int row = Mt + r + 8 * g;
        // res + wsp, summed over the A dimension (cols)
        rsum[r] += acc[r] + ob + bval[w][row] * wpwL[a] + wpbL[a];
      }
    }
    #pragma unroll
    for (int r = 0; r < 8; ++r) {
      float v = rsum[r];
      v += __shfl_xor(v, 1, 32);
      v += __shfl_xor(v, 2, 32);
      v += __shfl_xor(v, 4, 32);
      v += __shfl_xor(v, 8, 32);
      if (ln == 0) {
        int row = Mt + r + 8 * g;
        if (row < K_) dynacc[w][row] = v;
      }
    }
  }
  __syncthreads();
  if (lane < K_) {
    dyn[(size_t)((s * B_ + b) * C_ + c) * K_ + lane] = dynacc[w][lane] * (1.f / 64.f);
  }
}

// ---------------------------------------------------------------------------
// Kernel 4: fused dual strip conv + BN + GELU + gating.
// 32x32 output tile per block; 50x50 LDS tile serves both conv directions.
// Tile staging uses CDNA5 async global->LDS copies (ASYNCcnt-tracked),
// bypassing the register file; OOB halo elements are zero-filled with
// ordinary LDS stores (disjoint addresses, no ordering hazard).
// ---------------------------------------------------------------------------
__global__ __launch_bounds__(256)
void conv_fuse_kernel(const float* __restrict__ x, const float* __restrict__ dyn,
                      const float* __restrict__ bn_g, const float* __restrict__ bn_b,
                      const float* __restrict__ bn_m, const float* __restrict__ bn_v,
                      float* __restrict__ out) {
  __shared__ float tile[50 * 50];
  __shared__ float dk[2 * K_];
  int bid = blockIdx.x;
  const int txx = bid % 3; bid /= 3;
  const int tyy = bid % 3; bid /= 3;
  const int c = bid % C_;
  const int b = bid / C_;
  const int tid = threadIdx.x;
  const size_t plane = (size_t)(b * C_ + c) * (HW_ * HW_);
  const float* xp = x + plane;
  const int y0 = tyy * 32 - 9, x0 = txx * 32 - 9;

  for (int i = tid; i < 2500; i += 256) {
    int r = i / 50, col = i % 50;
    int gy = y0 + r, gx = x0 + col;
    if (gy >= 0 && gy < HW_ && gx >= 0 && gx < HW_) {
      // async global -> LDS copy (per-lane, EXEC-predicated), ASYNCcnt-tracked
      unsigned lds_addr = (unsigned)(uintptr_t)&tile[i];
      unsigned long long gaddr = (unsigned long long)(uintptr_t)(xp + gy * HW_ + gx);
      asm volatile("global_load_async_to_lds_b32 %0, %1, off"
                   :: "v"(lds_addr), "v"(gaddr)
                   : "memory");
    } else {
      tile[i] = 0.f;   // zero-pad halo
    }
  }
  if (tid < 2 * K_) {
    int s = tid / K_, k = tid % K_;
    dk[tid] = dyn[(size_t)((s * B_ + b) * C_ + c) * K_ + k];
  }
  const float scale = bn_g[c] * rsqrtf(bn_v[c] + 1e-5f);
  const float shift = bn_b[c] - bn_m[c] * scale;
  // drain this wave's async LDS writes before the block barrier
  asm volatile("s_wait_asynccnt 0x0" ::: "memory");
  __syncthreads();

  const int lx = tid & 31, ly0 = tid >> 5;
  #pragma unroll
  for (int rr = 0; rr < 4; ++rr) {
    int ly = ly0 + rr * 8;
    float oh = 0.f, ov = 0.f;
    #pragma unroll
    for (int k = 0; k < K_; ++k) {
      oh = fmaf(tile[(ly + 9) * 50 + lx + k], dk[k], oh);
      ov = fmaf(tile[(ly + k) * 50 + lx + 9], dk[K_ + k], ov);
    }
    float t  = (oh + ov) * scale + shift;
    float gl = gelu_exact(t);
    float res = tile[(ly + 9) * 50 + lx + 9] * gl;
    __builtin_nontemporal_store(
        res, &out[plane + (size_t)(tyy * 32 + ly) * HW_ + (txx * 32 + lx)]);
  }
}

// ---------------------------------------------------------------------------
extern "C" void kernel_launch(void* const* d_in, const int* in_sizes, int n_in,
                              void* d_out, int out_size, void* d_ws, size_t ws_size,
                              hipStream_t stream) {
  (void)in_sizes; (void)n_in; (void)out_size; (void)ws_size;
  const float* x      = (const float*)d_in[0];
  const float* w1     = (const float*)d_in[1];
  const float* b1     = (const float*)d_in[2];
  const float* w2     = (const float*)d_in[3];
  const float* b2     = (const float*)d_in[4];
  const float* wp_w   = (const float*)d_in[5];
  const float* wp_b   = (const float*)d_in[6];
  const float* in_w   = (const float*)d_in[7];
  const float* in_b   = (const float*)d_in[8];
  const float* out_w  = (const float*)d_in[9];
  const float* out_b  = (const float*)d_in[10];
  const float* base_h = (const float*)d_in[11];
  const float* base_v = (const float*)d_in[12];
  const float* bn_g   = (const float*)d_in[13];
  const float* bn_b   = (const float*)d_in[14];
  const float* bn_m   = (const float*)d_in[15];
  const float* bn_v   = (const float*)d_in[16];

  float* ws     = (float*)d_ws;
  float* pooled = ws;                    // B*C = 2048 floats
  float* modbuf = ws + 2048;             // B*3*64 = 1536 floats
  float* dynbuf = ws + 2048 + 1536;      // 2*B*C*K = 77824 floats

  pool_kernel<<<B_ * C_, 256, 0, stream>>>(x, pooled);
  mlp_kernel<<<1, 256, 0, stream>>>(pooled, w1, b1, w2, b2, modbuf);
  dyn_mha_kernel<<<(2 * B_ * C_) / 2, 64, 0, stream>>>(
      modbuf, wp_w, wp_b, in_w, in_b, out_w, out_b, base_h, base_v, dynbuf);
  conv_fuse_kernel<<<B_ * C_ * 9, 256, 0, stream>>>(
      x, dynbuf, bn_g, bn_b, bn_m, bn_v, (float*)d_out);
}
